// ExplainerGCMO_85040352461208
// MI455X (gfx1250) — matrix operand order repaired and checked by reading, not compiled, not run on hardware
//
#include <hip/hip_runtime.h>
#include <hip/hip_bf16.h>

// ---------------------------------------------------------------------------
// ExplainerGCMO for MI455X (gfx1250, wave32).
//   N=8192, E=262144, D=128, H=64
// Pipeline (all on `stream`):
//   1) zero_kernel:     d_out (N*N f32) <- 0       (d_out doubles as mask)
//   2) edge_mlp_kernel: bf16 WMMA edge MLP -> gate[e]; atomicAdd into d_out
//   3) finalize_kernel: in-place out = adj * 0.5*(mask + mask^T), tiled with
//                       LDS transpose staging; race-free by tile-pair ownership
// ---------------------------------------------------------------------------

typedef __bf16 v16bf __attribute__((ext_vector_type(16)));
typedef __bf16 v8bf  __attribute__((ext_vector_type(8)));
typedef float  v8f   __attribute__((ext_vector_type(8)));

union FragBF { v16bf v; v8bf h[2]; };

static constexpr int       kN = 8192;
static constexpr int       kD = 128;    // per-endpoint embed dim
static constexpr int       kK = 256;    // 2*D, GEMM K
static constexpr int       kH = 64;     // hidden
static constexpr long long kE = 262144;
static constexpr int       KPAD = 264;  // padded K stride in LDS (breaks bank conflicts)

__device__ __forceinline__ __bf16 to_bf16(float f) {
    unsigned u = __builtin_bit_cast(unsigned, f);
    u += 0x7FFFu + ((u >> 16) & 1u);          // round-to-nearest-even
    unsigned short s = (unsigned short)(u >> 16);
    return __builtin_bit_cast(__bf16, s);
}

// ---------------------------------------------------------------------------
// Kernel 1: zero the output/mask buffer (float4 grid-stride).
// ---------------------------------------------------------------------------
__global__ void zero_kernel(float4* __restrict__ out, int n4) {
    int i = blockIdx.x * blockDim.x + threadIdx.x;
    const int stride = gridDim.x * blockDim.x;
    const float4 z = make_float4(0.f, 0.f, 0.f, 0.f);
    for (; i < n4; i += stride) out[i] = z;
}

// ---------------------------------------------------------------------------
// Kernel 2: edge MLP with v_wmma_f32_16x16x32_bf16.
// Block = 256 threads = 8 waves; each wave owns 16 edges (block: 128 edges).
// GEMM tile per wave: M=16 edges x N=64 hidden, K=256 -> 8 kt x 4 nt WMMAs.
// ---------------------------------------------------------------------------
__global__ void __launch_bounds__(256)
edge_mlp_kernel(const float* __restrict__ embed,
                const float* __restrict__ W1,
                const float* __restrict__ b1,
                const float* __restrict__ W2,
                const float* __restrict__ b2,
                const float* __restrict__ noise,
                const long long* __restrict__ eidx,
                float* __restrict__ out /* mask accumulator, N*N */) {
    // LDS: W1 as bf16, N-major [64][KPAD]; per-wave A tiles [16][KPAD].
    __shared__ __bf16 sW1[kH * KPAD];            // ~33 KB
    __shared__ __bf16 sA[8 * 16 * KPAD];         // ~66 KB  (320 KB/WGP available)

    const int tid  = threadIdx.x;
    const int lane = tid & 31;
    const int wid  = tid >> 5;
    const int c    = lane & 15;    // column-in-tile / row-in-tile selector
    const int hh   = lane >> 4;    // lane half (K sub-range selector)

    // ---- stage W1 (256x64 f32, row-major) -> bf16, N-major ----
    for (int i = tid; i < kK * kH; i += 256) {
        int k = i >> 6;            // 0..255
        int n = i & 63;            // 0..63
        sW1[n * KPAD + k] = to_bf16(W1[i]);
    }

    // ---- gather+stage A: lane covers (edge eL, half) -> 128 floats -> bf16 ----
    const long long eBase = (long long)blockIdx.x * 128 + (long long)wid * 16;
    {
        const int eL   = c;                         // edge within wave tile
        const long long e = eBase + eL;
        const long long ridx = eidx[(long long)hh * kE + e];   // row / col endpoint
        const float4* src = (const float4*)(embed + ridx * kD);
        __bf16* dst = &sA[(wid * 16 + eL) * KPAD + hh * kD];
        #pragma unroll
        for (int q = 0; q < 16; ++q) {              // 16 x 8 floats = 128
            float4 f0 = src[q * 2 + 0];
            float4 f1 = src[q * 2 + 1];
            v8bf v;
            v[0] = to_bf16(f0.x); v[1] = to_bf16(f0.y);
            v[2] = to_bf16(f0.z); v[3] = to_bf16(f0.w);
            v[4] = to_bf16(f1.x); v[5] = to_bf16(f1.y);
            v[6] = to_bf16(f1.z); v[7] = to_bf16(f1.w);
            *(v8bf*)(dst + q * 8) = v;              // 16B-aligned ds_store_b128
        }
    }
    __syncthreads();

    // ---- per-lane constants for epilogue ----
    float bias[4], w2v[4];
    #pragma unroll
    for (int nt = 0; nt < 4; ++nt) {
        bias[nt] = b1[nt * 16 + c];
        w2v[nt]  = W2[nt * 16 + c];
    }

    v8f acc[4];
    #pragma unroll
    for (int nt = 0; nt < 4; ++nt)
        #pragma unroll
        for (int v = 0; v < 8; ++v) acc[nt][v] = 0.f;

    // ---- main WMMA loop: K = 256 in 8 steps of 32 ----
    // A frag (contiguous k order, same as B): lane l holds row M=l%16,
    //   element e -> K = kt*32 + hh*16 + e
    // B frag (contiguous k order, same as A): lane l holds col N=l%16,
    //   element e -> K = kt*32 + hh*16 + e
    const __bf16* aBase = &sA[(wid * 16 + c) * KPAD];
    #pragma unroll
    for (int kt = 0; kt < 8; ++kt) {
        FragBF a;
        a.h[0] = *(const v8bf*)(aBase + kt * 32 + hh * 16);
        a.h[1] = *(const v8bf*)(aBase + kt * 32 + hh * 16 + 8);
        #pragma unroll
        for (int nt = 0; nt < 4; ++nt) {
            FragBF b;
            const __bf16* bBase = &sW1[(nt * 16 + c) * KPAD + kt * 32 + hh * 16];
            b.h[0] = *(const v8bf*)(bBase + 0);
            b.h[1] = *(const v8bf*)(bBase + 8);
            acc[nt] = __builtin_amdgcn_wmma_f32_16x16x32_bf16(
                false, a.v, false, b.v, (short)0, acc[nt], false, false);
        }
    }

    // ---- epilogue: relu(acc + b1) . W2, reduce over N=64 ----
    // C layout: lanes 0-15 VGPR v -> row M=v; lanes 16-31 -> row M=v+8.
    float p[8];
    #pragma unroll
    for (int v = 0; v < 8; ++v) {
        float s = 0.f;
        #pragma unroll
        for (int nt = 0; nt < 4; ++nt) {
            float x = acc[nt][v] + bias[nt];
            x = x > 0.f ? x : 0.f;
            s += x * w2v[nt];
        }
        p[v] = s;
    }
    // reduce across the 16 lanes of each half-wave
    #pragma unroll
    for (int m = 1; m < 16; m <<= 1)
        #pragma unroll
        for (int v = 0; v < 8; ++v) p[v] += __shfl_xor(p[v], m, 16);

    // lanes 0 and 16 own rows 0..7 / 8..15 of the wave's 16-edge tile
    if (c == 0) {
        const float bb = b2[0];
        #pragma unroll
        for (int v = 0; v < 8; ++v) {
            const long long e  = eBase + hh * 8 + v;
            const float la = p[v] + bb;
            const float u  = noise[e];
            const float z  = logf(u) - log1pf(-u) + la;
            const float g  = 1.f / (1.f + expf(-z));
            const long long r  = eidx[e];
            const long long cl = eidx[kE + e];
            atomicAdd(&out[r * (long long)kN + cl], g);  // coalesce duplicates
        }
    }
}

// ---------------------------------------------------------------------------
// Kernel 3: in-place  out[i][j] = adj[i][j] * 0.5*(mask[i][j] + mask[j][i]).
// 64x64 tile-pairs; block (bi,bj) with bi<=bj owns tiles (bi,bj) and (bj,bi):
// stage both into LDS (coalesced), barrier, then write both (transpose via LDS).
// ---------------------------------------------------------------------------
__global__ void __launch_bounds__(256)
finalize_kernel(const float* __restrict__ adj, float* __restrict__ out) {
    const int bi = blockIdx.y, bj = blockIdx.x;
    if (bi > bj) return;

    __shared__ float mA[64][65];
    __shared__ float mB[64][65];

    const int t  = threadIdx.x;
    const int cc = t & 63;
    const int r0 = t >> 6;                 // 0..3
    const long long gi = (long long)bi * 64, gj = (long long)bj * 64;

    #pragma unroll 4
    for (int p = 0; p < 16; ++p) {
        const int rr = r0 + p * 4;
        mA[rr][cc] = out[(gi + rr) * kN + gj + cc];
        if (bi != bj) mB[rr][cc] = out[(gj + rr) * kN + gi + cc];
    }
    __syncthreads();

    #pragma unroll 4
    for (int p = 0; p < 16; ++p) {
        const int rr = r0 + p * 4;
        if (bi == bj) {
            const float s = 0.5f * (mA[rr][cc] + mA[cc][rr]);
            out[(gi + rr) * kN + gj + cc] = adj[(gi + rr) * kN + gj + cc] * s;
        } else {
            const float s1 = 0.5f * (mA[rr][cc] + mB[cc][rr]);
            out[(gi + rr) * kN + gj + cc] = adj[(gi + rr) * kN + gj + cc] * s1;
            const float s2 = 0.5f * (mB[rr][cc] + mA[cc][rr]);
            out[(gj + rr) * kN + gi + cc] = adj[(gj + rr) * kN + gi + cc] * s2;
        }
    }
}

// ---------------------------------------------------------------------------
extern "C" void kernel_launch(void* const* d_in, const int* in_sizes, int n_in,
                              void* d_out, int out_size, void* d_ws, size_t ws_size,
                              hipStream_t stream) {
    const float*     embed = (const float*)d_in[0];
    const float*     adj   = (const float*)d_in[1];
    const float*     noise = (const float*)d_in[2];
    const float*     W1    = (const float*)d_in[3];
    const float*     b1    = (const float*)d_in[4];
    const float*     W2    = (const float*)d_in[5];
    const float*     b2    = (const float*)d_in[6];
    const long long* eidx  = (const long long*)d_in[7];
    float*           out   = (float*)d_out;

    // 1) zero mask/output (poisoned by harness)
    const int n4 = (kN * kN) / 4;
    zero_kernel<<<4096, 256, 0, stream>>>((float4*)out, n4);

    // 2) WMMA edge MLP + gate scatter: E/128 blocks of 8 waves x 16 edges
    const int mlpBlocks = (int)(kE / 128);
    edge_mlp_kernel<<<mlpBlocks, 256, 0, stream>>>(embed, W1, b1, W2, b2,
                                                   noise, eidx, out);

    // 3) symmetrize * adj, in place, 64x64 tile pairs
    dim3 fgrid(kN / 64, kN / 64);
    finalize_kernel<<<fgrid, 256, 0, stream>>>(adj, out);
}
